// MultiheadFlashDiff1_25288767439593
// MI455X (gfx1250) — matrix operand order repaired and checked by reading, not compiled, not run on hardware
//
#include <hip/hip_runtime.h>
#include <hip/hip_bf16.h>
#include <cstdint>

typedef __bf16 bf16;
typedef __attribute__((ext_vector_type(16))) __bf16 bf16x16;
typedef __attribute__((ext_vector_type(8)))  __bf16 bf16x8;
typedef __attribute__((ext_vector_type(4)))  __bf16 bf16x4;
typedef __attribute__((ext_vector_type(8)))  float  f32x8;

#define B_ 2
#define T_ 2048
#define E_ 2048
#define H_ 16
#define D_ 64
#define LAMBDA_INIT_F 0.783605766531624f   // 0.8 - 0.6*exp(-0.3*12)

// ---------------------------------------------------------------------------
// DPP butterfly reductions across the 16 lanes of a wave32 half.
// ---------------------------------------------------------------------------
__device__ __forceinline__ float red_max16(float x) {
  x = fmaxf(x, __int_as_float(__builtin_amdgcn_update_dpp(0, __float_as_int(x), 0xB1,  0xf, 0xf, true)));
  x = fmaxf(x, __int_as_float(__builtin_amdgcn_update_dpp(0, __float_as_int(x), 0x4E,  0xf, 0xf, true)));
  x = fmaxf(x, __int_as_float(__builtin_amdgcn_update_dpp(0, __float_as_int(x), 0x141, 0xf, 0xf, true)));
  x = fmaxf(x, __int_as_float(__builtin_amdgcn_update_dpp(0, __float_as_int(x), 0x140, 0xf, 0xf, true)));
  return x;
}
__device__ __forceinline__ float red_sum16(float x) {
  x += __int_as_float(__builtin_amdgcn_update_dpp(0, __float_as_int(x), 0xB1,  0xf, 0xf, true));
  x += __int_as_float(__builtin_amdgcn_update_dpp(0, __float_as_int(x), 0x4E,  0xf, 0xf, true));
  x += __int_as_float(__builtin_amdgcn_update_dpp(0, __float_as_int(x), 0x141, 0xf, 0xf, true));
  x += __int_as_float(__builtin_amdgcn_update_dpp(0, __float_as_int(x), 0x140, 0xf, 0xf, true));
  return x;
}
// value held by lane^1 (rotation partner in the C-fragment layout)
__device__ __forceinline__ float dpp_swap1(float x) {
  return __int_as_float(__builtin_amdgcn_update_dpp(0, __float_as_int(x), 0xB1, 0xf, 0xf, true));
}
// generic pointer -> 32-bit LDS offset (for async-to-LDS instructions)
__device__ __forceinline__ unsigned lds_off_u32(void* p) {
  return (unsigned)(uintptr_t)(__attribute__((address_space(3))) void*)p;
}

// ---------------------------------------------------------------------------
// fp32 -> bf16 conversion (x4 vectorized)
// ---------------------------------------------------------------------------
__global__ void diffattn_cvt_bf16(const float* __restrict__ src,
                                  bf16* __restrict__ dst, int n4) {
  int i = blockIdx.x * blockDim.x + threadIdx.x;
  if (i < n4) {
    float4 v = ((const float4*)src)[i];
    bf16x4 o;
    o[0] = (bf16)v.x; o[1] = (bf16)v.y; o[2] = (bf16)v.z; o[3] = (bf16)v.w;
    ((bf16x4*)dst)[i] = o;
  }
}

// ---------------------------------------------------------------------------
// C = A(MxK) @ W(NxK)^T  via v_wmma_f32_16x16x32_bf16.
// Block = 128 threads (4 waves); wave computes 32x64 of C.
// MODE 0: bf16 store; MODE 1: bf16 store + fused interleaved RoPE (DPP pair
// swap: rotation pairs are adjacent lanes in the C layout); MODE 2: f32 store.
// ---------------------------------------------------------------------------
template<int MODE>
__global__ __launch_bounds__(128)
void diffattn_gemm_bf16(const bf16* __restrict__ A, const bf16* __restrict__ W,
                        void* __restrict__ Cout, int M, int N, int K,
                        const float* __restrict__ cosp, const float* __restrict__ sinp) {
  const int wave = threadIdx.x >> 5;
  const int lane = threadIdx.x & 31;
  const int hi   = lane >> 4;
  const int l15  = lane & 15;
  const int m0   = (blockIdx.y * 4 + wave) * 32;
  const int n0   = blockIdx.x * 64;

  f32x8 acc[2][4];
  for (int a = 0; a < 2; ++a)
    for (int j = 0; j < 4; ++j)
      for (int e = 0; e < 8; ++e) acc[a][j][e] = 0.0f;

  const int akb = hi ? 8 : 0;
  const int bkb = hi ? 8 : 0;
  const bf16* arow0 = A + (size_t)(m0 + l15) * K + akb;
  const bf16* arow1 = A + (size_t)(m0 + 16 + l15) * K + akb;

  for (int k0 = 0; k0 < K; k0 += 32) {
    if (k0 + 256 < K) {
      __builtin_prefetch(arow0 + k0 + 256, 0, 1);
      __builtin_prefetch(arow1 + k0 + 256, 0, 1);
    }
    bf16x16 afrag[2];
    #pragma unroll
    for (int a = 0; a < 2; ++a) {
      const bf16* ap = (a == 0 ? arow0 : arow1) + k0;
      bf16x8 alo = *(const bf16x8*)ap;
      bf16x8 ahi = *(const bf16x8*)(ap + 16);
      #pragma unroll
      for (int i = 0; i < 8; ++i) { afrag[a][i] = alo[i]; afrag[a][i + 8] = ahi[i]; }
    }
    #pragma unroll
    for (int j = 0; j < 4; ++j) {
      const bf16* bp = W + (size_t)(n0 + j * 16 + l15) * K + k0 + bkb;
      bf16x8 blo = *(const bf16x8*)bp;
      bf16x8 bhi = *(const bf16x8*)(bp + 16);
      bf16x16 bfrag;
      #pragma unroll
      for (int i = 0; i < 8; ++i) { bfrag[i] = blo[i]; bfrag[i + 8] = bhi[i]; }
      acc[0][j] = __builtin_amdgcn_wmma_f32_16x16x32_bf16(
          false, afrag[0], false, bfrag, (short)0, acc[0][j], false, false);
      acc[1][j] = __builtin_amdgcn_wmma_f32_16x16x32_bf16(
          false, afrag[1], false, bfrag, (short)0, acc[1][j], false, false);
    }
  }

  const int odd = l15 & 1;
  #pragma unroll
  for (int a = 0; a < 2; ++a) {
    #pragma unroll
    for (int j = 0; j < 4; ++j) {
      const int n = n0 + j * 16 + l15;
      #pragma unroll
      for (int r = 0; r < 8; ++r) {
        const int m = m0 + a * 16 + r + (hi ? 8 : 0);
        float v = acc[a][j][r];
        if (MODE == 1) {   // fused RoPE: partner value lives in lane^1
          float p = dpp_swap1(v);
          int   t = m & (T_ - 1);
          int   i = (n & 63) >> 1;
          float c = cosp[t * 32 + i];
          float s = sinp[t * 32 + i];
          v = odd ? (p * s + v * c) : (v * c - p * s);
        }
        if (MODE == 2) ((float*)Cout)[(size_t)m * N + n] = v;
        else           ((bf16*) Cout)[(size_t)m * N + n] = (bf16)v;
      }
    }
  }
}

// ---------------------------------------------------------------------------
// lambda_full = exp(sum lq1*lk1) - exp(sum lq2*lk2) + LAMBDA_INIT  (one wave)
// ---------------------------------------------------------------------------
__global__ void diffattn_lambda(const float* __restrict__ lq1, const float* __restrict__ lk1,
                                const float* __restrict__ lq2, const float* __restrict__ lk2,
                                float* __restrict__ lam) {
  int l = threadIdx.x;
  float s1 = lq1[l] * lk1[l] + lq1[l + 32] * lk1[l + 32];
  float s2 = lq2[l] * lk2[l] + lq2[l + 32] * lk2[l + 32];
  s1 = red_sum16(s1); s1 += __shfl_xor(s1, 16, 32);
  s2 = red_sum16(s2); s2 += __shfl_xor(s2, 16, 32);
  if (l == 0) lam[0] = __expf(s1) - __expf(s2) + LAMBDA_INIT_F;
}

// ---------------------------------------------------------------------------
// Fused differential flash attention + combine for one (b, head, 64-row qtile).
// 256 threads = 8 waves: waves 0-3 compute component 0 (h2=2h), waves 4-7
// component 1 (h2=2h+1); the V tile is shared. K tiles staged with
// global_load_async_to_lds_b128 (ASYNCcnt). After the K loop, comp-1 waves
// export normalized rows through LDS; comp-0 waves apply a1 - lam*a2,
// RMS-norm(128) * g * (1-LAMBDA_INIT) and store bf16 directly to ab.
// ---------------------------------------------------------------------------
__global__ __launch_bounds__(256)
void diffattn_flash(const bf16* __restrict__ qb, const bf16* __restrict__ kbuf,
                    const bf16* __restrict__ vbuf, const float* __restrict__ g,
                    const float* __restrict__ lamp, bf16* __restrict__ ab) {
  __shared__ __align__(16) bf16 Kt[2][32 * 72];   // per-component K tile, +8 pad
  __shared__ __align__(16) bf16 Vt[128 * 40];     // shared V^T tile, +8 pad
  __shared__ __align__(16) bf16 Pt[8][16 * 40];   // per-wave P tile, +8 pad
  __shared__ __align__(16) float Xt[4][16][128];  // attn2 exchange (32 KB)

  const int b    = blockIdx.z;
  const int head = blockIdx.y;
  const int qt   = blockIdx.x;
  const int wave = threadIdx.x >> 5;
  const int lane = threadIdx.x & 31;
  const int hi   = lane >> 4;
  const int l15  = lane & 15;
  const int comp = wave >> 2;          // differential component
  const int wq   = wave & 3;           // q-row subgroup
  const int h2   = head * 2 + comp;
  const int qbase = qt * 64 + wq * 16;
  const float scale = 0.125f;          // D^-0.5

  // Q fragments (16 rows x 64 dims -> two 16x32 A-frags)
  const int akb = hi ? 8 : 0;
  const bf16* qrow = qb + ((size_t)(b * T_ + qbase + l15) * E_) + h2 * 64 + akb;
  bf16x16 qa[2];
  #pragma unroll
  for (int f = 0; f < 2; ++f) {
    bf16x8 lo = *(const bf16x8*)(qrow + f * 32);
    bf16x8 hh = *(const bf16x8*)(qrow + f * 32 + 16);
    #pragma unroll
    for (int i = 0; i < 8; ++i) { qa[f][i] = lo[i]; qa[f][i + 8] = hh[i]; }
  }

  float mrow[8], lrow[8];
  f32x8 acc[8];
  #pragma unroll
  for (int r = 0; r < 8; ++r) { mrow[r] = -3.0e38f; lrow[r] = 0.0f; }
  for (int j = 0; j < 8; ++j)
    for (int e = 0; e < 8; ++e) acc[j][e] = 0.0f;

  // staging assignments
  const int kcomp = threadIdx.x >> 7;        // which component's K tile
  const int krr   = threadIdx.x & 127;
  const int kkey  = krr >> 2, kch = krr & 3; // 32 keys x 4 chunks of 16 halfs
  const int vkey  = threadIdx.x >> 3;        // 32 keys x 8 chunks of 16 halfs
  const int vch   = threadIdx.x & 7;

  const int kend = qt * 64 + 64;
  for (int kt = 0; kt < kend; kt += 32) {
    __syncthreads();   // all waves done reading previous tiles
    {
      uint64_t ga = (uint64_t)(kbuf + ((size_t)(b * T_ + kt + kkey) * E_)
                               + (head * 2 + kcomp) * 64 + kch * 16);
      unsigned la = lds_off_u32(&Kt[kcomp][kkey * 72 + kch * 16]);
      asm volatile("global_load_async_to_lds_b128 %0, %1, off"
                   :: "v"(la), "v"(ga) : "memory");
      asm volatile("global_load_async_to_lds_b128 %0, %1, off offset:16"
                   :: "v"(la), "v"(ga) : "memory");
      const bf16* vp = vbuf + ((size_t)(b * T_ + kt + vkey) * E_) + head * 128 + vch * 16;
      #pragma unroll
      for (int j = 0; j < 16; ++j) Vt[(vch * 16 + j) * 40 + vkey] = vp[j];
    }
    asm volatile("s_wait_asynccnt 0x0" ::: "memory");
    __syncthreads();

    // S = Q x K^T : two 16x16 fp32 fragments (key groups 0-15, 16-31)
    f32x8 sc[2];
    #pragma unroll
    for (int gi = 0; gi < 2; ++gi) {
      for (int e = 0; e < 8; ++e) sc[gi][e] = 0.0f;
      #pragma unroll
      for (int f = 0; f < 2; ++f) {
        const bf16* kpl = &Kt[comp][(gi * 16 + l15) * 72 + f * 32 + (hi ? 8 : 0)];
        bf16x8 lo = *(const bf16x8*)kpl;
        bf16x8 hh = *(const bf16x8*)(kpl + 16);
        bf16x16 kfrag;
        #pragma unroll
        for (int i = 0; i < 8; ++i) { kfrag[i] = lo[i]; kfrag[i + 8] = hh[i]; }
        sc[gi] = __builtin_amdgcn_wmma_f32_16x16x32_bf16(
            false, qa[f], false, kfrag, (short)0, sc[gi], false, false);
      }
    }

    // online softmax per row
    #pragma unroll
    for (int r = 0; r < 8; ++r) {
      int grow = qbase + r + hi * 8;
      float s0 = sc[0][r] * scale; if (kt + l15 > grow)      s0 = -3.0e38f;
      float s1 = sc[1][r] * scale; if (kt + 16 + l15 > grow) s1 = -3.0e38f;
      float tm    = red_max16(fmaxf(s0, s1));
      float mnew  = fmaxf(mrow[r], tm);
      float alpha = __expf(mrow[r] - mnew);
      float p0 = __expf(s0 - mnew);
      float p1 = __expf(s1 - mnew);
      float ts = red_sum16(p0 + p1);
      lrow[r] = lrow[r] * alpha + ts;
      mrow[r] = mnew;
      #pragma unroll
      for (int j = 0; j < 8; ++j) acc[j][r] *= alpha;
      Pt[wave][(r + hi * 8) * 40 + l15]      = (bf16)p0;
      Pt[wave][(r + hi * 8) * 40 + 16 + l15] = (bf16)p1;
    }
    // Pt is wave-private; DS ops are in-order per wave -> wave-local wait only.
    asm volatile("s_wait_dscnt 0x0" ::: "memory");

    const bf16* pp = &Pt[wave][l15 * 40 + (hi ? 8 : 0)];
    bf16x8 plo = *(const bf16x8*)pp;
    bf16x8 phi = *(const bf16x8*)(pp + 16);
    bf16x16 pfrag;
    #pragma unroll
    for (int i = 0; i < 8; ++i) { pfrag[i] = plo[i]; pfrag[i + 8] = phi[i]; }

    #pragma unroll
    for (int j = 0; j < 8; ++j) {
      const bf16* vpl = &Vt[(j * 16 + l15) * 40 + (hi ? 8 : 0)];
      bf16x8 vlo = *(const bf16x8*)vpl;
      bf16x8 vhi = *(const bf16x8*)(vpl + 16);
      bf16x16 vfrag;
      #pragma unroll
      for (int i = 0; i < 8; ++i) { vfrag[i] = vlo[i]; vfrag[i + 8] = vhi[i]; }
      acc[j] = __builtin_amdgcn_wmma_f32_16x16x32_bf16(
          false, pfrag, false, vfrag, (short)0, acc[j], false, false);
    }
  }

  // --- differential combine: a1 - lam*a2, RMS-norm, *g*(1-li), bf16 store ---
  if (comp == 1) {
    #pragma unroll
    for (int r = 0; r < 8; ++r) {
      float inv = 1.0f / lrow[r];
      #pragma unroll
      for (int j = 0; j < 8; ++j)
        Xt[wq][r + hi * 8][j * 16 + l15] = acc[j][r] * inv;
    }
  }
  __syncthreads();
  if (comp == 0) {
    float lam = lamp[0];
    float gj[8];
    #pragma unroll
    for (int j = 0; j < 8; ++j) gj[j] = g[j * 16 + l15];
    #pragma unroll
    for (int r = 0; r < 8; ++r) {
      float inv = 1.0f / lrow[r];
      float av[8];
      float ss = 0.0f;
      #pragma unroll
      for (int j = 0; j < 8; ++j) {
        float a2 = Xt[wq][r + hi * 8][j * 16 + l15];
        av[j] = acc[j][r] * inv - lam * a2;
        ss += av[j] * av[j];
      }
      ss = red_sum16(ss);
      float rn = rsqrtf(ss * (1.0f / 128.0f) + 1e-5f) * (1.0f - LAMBDA_INIT_F);
      int grow = qbase + r + hi * 8;
      size_t ob = ((size_t)(b * T_ + grow) * E_) + head * 128 + l15;
      #pragma unroll
      for (int j = 0; j < 8; ++j)
        ab[ob + j * 16] = (bf16)(av[j] * rn * gj[j]);
    }
  }
}

// ---------------------------------------------------------------------------
extern "C" void kernel_launch(void* const* d_in, const int* in_sizes, int n_in,
                              void* d_out, int out_size, void* d_ws, size_t ws_size,
                              hipStream_t stream) {
  (void)in_sizes; (void)n_in; (void)out_size; (void)ws_size;
  const float* x    = (const float*)d_in[0];
  const float* cosp = (const float*)d_in[1];
  const float* sinp = (const float*)d_in[2];
  const float* Wq   = (const float*)d_in[3];
  const float* Wk   = (const float*)d_in[4];
  const float* Wv   = (const float*)d_in[5];
  const float* Wo   = (const float*)d_in[6];
  const float* lq1  = (const float*)d_in[7];
  const float* lk1  = (const float*)d_in[8];
  const float* lq2  = (const float*)d_in[9];
  const float* lk2  = (const float*)d_in[10];
  const float* g    = (const float*)d_in[11];
  float* out = (float*)d_out;

  const size_t BTE = (size_t)B_ * T_ * E_;   // 8388608
  const size_t EE  = (size_t)E_ * E_;        // 4194304
  char* w = (char*)d_ws;
  bf16* xb   = (bf16*)w; w += BTE * 2;
  bf16* wqb  = (bf16*)w; w += EE * 2;
  bf16* wkb  = (bf16*)w; w += EE * 2;
  bf16* wvb  = (bf16*)w; w += EE * 2;
  bf16* wob  = (bf16*)w; w += EE * 2;
  bf16* qb   = (bf16*)w; w += BTE * 2;
  bf16* kbuf = (bf16*)w; w += BTE * 2;
  bf16* vbuf = (bf16*)w; w += BTE * 2;
  bf16* ab   = (bf16*)w; w += BTE * 2;
  float* lam = (float*)w; w += 256;

  diffattn_cvt_bf16<<<(int)(BTE / 1024), 256, 0, stream>>>(x,  xb,  (int)(BTE / 4));
  diffattn_cvt_bf16<<<(int)(EE  / 1024), 256, 0, stream>>>(Wq, wqb, (int)(EE / 4));
  diffattn_cvt_bf16<<<(int)(EE  / 1024), 256, 0, stream>>>(Wk, wkb, (int)(EE / 4));
  diffattn_cvt_bf16<<<(int)(EE  / 1024), 256, 0, stream>>>(Wv, wvb, (int)(EE / 4));
  diffattn_cvt_bf16<<<(int)(EE  / 1024), 256, 0, stream>>>(Wo, wob, (int)(EE / 4));

  dim3 ggrid(E_ / 64, (B_ * T_) / 128);   // (32, 32), 32x64 tile per wave
  // q/k projections with fused interleaved RoPE in the epilogue
  diffattn_gemm_bf16<1><<<ggrid, 128, 0, stream>>>(xb, wqb, qb,   B_ * T_, E_, E_, cosp, sinp);
  diffattn_gemm_bf16<1><<<ggrid, 128, 0, stream>>>(xb, wkb, kbuf, B_ * T_, E_, E_, cosp, sinp);
  diffattn_gemm_bf16<0><<<ggrid, 128, 0, stream>>>(xb, wvb, vbuf, B_ * T_, E_, E_, cosp, sinp);

  diffattn_lambda<<<1, 32, 0, stream>>>(lq1, lk1, lq2, lk2, lam);

  // fused differential flash attention + combine
  diffattn_flash<<<dim3(T_ / 64, H_, B_), 256, 0, stream>>>(qb, kbuf, vbuf, g, lam, ab);

  // output projection
  diffattn_gemm_bf16<2><<<ggrid, 128, 0, stream>>>(ab, wob, (void*)out, B_ * T_, E_, E_, cosp, sinp);
}
